// Graphcl_29059748725495
// MI455X (gfx1250) — compile-verified, matrix operand order checked by reading, not run on hardware
//
#include <hip/hip_runtime.h>

// ---------------- problem constants (from reference) ----------------
#define N_NODES 50000
#define N_EDGES 500000
#define N_ROWTILES (N_NODES / 16)   // 3125, exact

// ---------------- WMMA types (CDNA5 / gfx1250, wave32) ----------------
typedef __attribute__((ext_vector_type(16))) __bf16 v16bf;
typedef __attribute__((ext_vector_type(8)))  float  v8f;

union BF16Frag { v16bf v; unsigned short s[16]; };

__device__ __forceinline__ unsigned short f2bf(float f) {
  // round-to-nearest-even fp32 -> bf16 bit pattern
  unsigned u = __float_as_uint(f);
  u += 0x7FFFu + ((u >> 16) & 1u);
  return (unsigned short)(u >> 16);
}

__device__ __forceinline__ float gelu_exact(float x) {
  return 0.5f * x * (1.0f + erff(x * 0.70710678118654752f));
}

// float atomic max via sign-split int atomics (works with -FLT_MAX init)
__device__ __forceinline__ void atomicMaxF(float* addr, float val) {
  if (val >= 0.0f) atomicMax((int*)addr, __float_as_int(val));
  else             atomicMin((unsigned int*)addr, (unsigned int)__float_as_int(val));
}

// =====================================================================
// Generic WMMA GEMM:  out[N x C] = epilogue( X[N x K] @ W[K x C] + bias )
//   Fully specialized at compile time:
//     KC    : K/32 chunks,  C : output cols (64 or 128)
//     MODE  : 0 plain, 1 gate*sigmoid(y+bias), 2 leaky_relu(y+bias, 0.01)
//     HASB  : bias present,  SPLIT : A cols [0,KA) from XA, rest from XB
//   W staged into LDS as bf16 pre-swizzled to WMMA B-fragment order so a
//   B fragment is one contiguous 32-byte LDS vector load.
// =====================================================================
template <int KC, int C, int MODE, bool HASB, bool SPLIT>
__global__ void __launch_bounds__(256)
wmma_gemm_kernel(const float* __restrict__ XA, int ldxa, int KA,
                 const float* __restrict__ XB, int ldxb,
                 const float* __restrict__ W,
                 const float* __restrict__ bias,
                 float* __restrict__ out, int ldo,
                 int nRowTiles,
                 const float* __restrict__ gate, int ldg)
{
  constexpr int K       = KC * 32;
  constexpr int CT      = C >> 4;
  constexpr int ctShift = (C == 64) ? 2 : 3;      // log2(CT)
  __shared__ unsigned short Wlds[K * C];

  // ---- stage W (K x C, row-major f32) -> LDS bf16 in B-fragment order.
  // B 32x16 ISA layout: lane<16 -> K={0..7,16..23}, lane>=16 -> K={8..15,24..31};
  // n = ct*16 + (lane&15).
  for (int i = threadIdx.x; i < K * C; i += 256) {
    const int e  = i & 15;
    const int l  = (i >> 4) & 31;
    const int f  = i >> 9;
    const int ct = f & (CT - 1);
    const int kc = f >> ctShift;
    const int n  = (ct << 4) + (l & 15);
    const int b0 = (kc << 5) + (((l >> 4) & 1) << 3);
    const int k  = b0 + ((e < 8) ? e : (8 + e));   // e>=8 -> b0+16+(e-8)
    Wlds[i] = f2bf(W[k * C + n]);
  }
  __syncthreads();

  const int lane   = threadIdx.x & 31;
  const int wave   = blockIdx.x * 8 + (threadIdx.x >> 5);
  const int nWaves = gridDim.x * 8;
  const int khalf8 = ((lane >> 4) & 1) * 8;
  const int nsub   = lane & 15;

  for (int rt = wave; rt < nRowTiles; rt += nWaves) {  // wave-uniform: EXEC all 1s
    const int r0   = rt << 4;
    const int arow = r0 + nsub;

    // ---- A fragments (16x32 bf16 per K-chunk): two 8-wide K groups per
    // lane, each 32B-aligned -> float4 global loads + bf16 pack.
    v16bf afrag[KC];
#pragma unroll
    for (int kc = 0; kc < KC; ++kc) {
      BF16Frag u;
      const int b0 = kc * 32 + khalf8;
      {
        const float* p;
        if (SPLIT) p = (b0 < KA) ? (XA + arow * ldxa + b0)
                                 : (XB + arow * ldxb + (b0 - KA));
        else       p = XA + arow * ldxa + b0;
        const float4 x0 = *(const float4*)(p);
        const float4 x1 = *(const float4*)(p + 4);
        u.s[0] = f2bf(x0.x); u.s[1] = f2bf(x0.y); u.s[2] = f2bf(x0.z); u.s[3] = f2bf(x0.w);
        u.s[4] = f2bf(x1.x); u.s[5] = f2bf(x1.y); u.s[6] = f2bf(x1.z); u.s[7] = f2bf(x1.w);
      }
      {
        const int k0 = b0 + 16;
        const float* p;
        if (SPLIT) p = (k0 < KA) ? (XA + arow * ldxa + k0)
                                 : (XB + arow * ldxb + (k0 - KA));
        else       p = XA + arow * ldxa + k0;
        const float4 x0 = *(const float4*)(p);
        const float4 x1 = *(const float4*)(p + 4);
        u.s[8]  = f2bf(x0.x); u.s[9]  = f2bf(x0.y); u.s[10] = f2bf(x0.z); u.s[11] = f2bf(x0.w);
        u.s[12] = f2bf(x1.x); u.s[13] = f2bf(x1.y); u.s[14] = f2bf(x1.z); u.s[15] = f2bf(x1.w);
      }
      afrag[kc] = u.v;
    }

#pragma unroll
    for (int ct = 0; ct < CT; ++ct) {
      const int col = (ct << 4) + nsub;
      v8f acc = {0.f, 0.f, 0.f, 0.f, 0.f, 0.f, 0.f, 0.f};
#pragma unroll
      for (int kc = 0; kc < KC; ++kc) {
        // one contiguous 32-byte LDS vector load per B fragment
        const v16bf bfrag = *(const v16bf*)(
            Wlds + (((((kc << ctShift) + ct) << 5) + lane) << 4));
        acc = __builtin_amdgcn_wmma_f32_16x16x32_bf16(
            false, afrag[kc], false, bfrag, (short)0, acc, false, false);
      }
      // C/D layout: VGPR v -> m = v + 8*(lane>=16), n = lane&15
      const int mbase = r0 + ((lane >> 4) << 3);
      const float bv  = HASB ? bias[col] : 0.0f;
      float* po = out + mbase * ldo + col;
      const float* pg = (MODE == 1) ? (gate + mbase * ldg + col) : nullptr;
#pragma unroll
      for (int v = 0; v < 8; ++v) {
        float y = acc[v] + bv;
        if (MODE == 1)      y = pg[v * ldg] / (1.0f + __expf(-y));  // gate * sigmoid
        else if (MODE == 2) y = (y > 0.0f) ? y : 0.01f * y;         // leaky 0.01
        po[v * ldo] = y;
      }
    }
  }
}

// ---------------- elementwise / graph kernels ----------------
__global__ void fill_kernel(float* p, float v, int n) {
  int i = blockIdx.x * blockDim.x + threadIdx.x;
  if (i < n) p[i] = v;
}

__global__ void deg_kernel(const int* __restrict__ dst, float* __restrict__ deg, int E) {
  int e = blockIdx.x * blockDim.x + threadIdx.x;
  if (e < E) atomicAdd(&deg[dst[e]], 1.0f);
}

// agg[dst] += Y[src] over 64 dims (projected-first SAGE aggregation)
__global__ void scatter_add64_kernel(const float* __restrict__ Y,
                                     const int* __restrict__ src,
                                     const int* __restrict__ dst,
                                     float* __restrict__ agg, int E) {
  int t = blockIdx.x * blockDim.x + threadIdx.x;
  if (t >= E * 16) return;
  int e = t >> 4, g = (t & 15) << 2;
  float4 v = *(const float4*)(Y + src[e] * 64 + g);
  float* a = agg + dst[e] * 64 + g;
  atomicAdd(a + 0, v.x); atomicAdd(a + 1, v.y);
  atomicAdd(a + 2, v.z); atomicAdd(a + 3, v.w);
}

__global__ void sage_combine_kernel(const float* __restrict__ Yself,
                                    const float* __restrict__ agg,
                                    const float* __restrict__ deg,
                                    const float* __restrict__ b,
                                    const float* __restrict__ drop,
                                    float* __restrict__ out, int n) {
  int i = blockIdx.x * blockDim.x + threadIdx.x;
  if (i >= n) return;
  int row = i >> 6, d = i & 63;
  float dg = deg[row]; dg = dg > 1.0f ? dg : 1.0f;
  float x  = Yself[i] + agg[i] / dg + b[d];
  float g  = gelu_exact(x);
  if (drop) g *= drop[i];
  out[i] = g;
}

// logits[e,h] = attn[h] . leaky_relu(fs[src]+fd[dst], 0.2);  m[dst,h] = max
__global__ void gat_logits_kernel(const float* __restrict__ fs,
                                  const float* __restrict__ fd,
                                  const float* __restrict__ attn,
                                  const int* __restrict__ src,
                                  const int* __restrict__ dst,
                                  float* __restrict__ logits,
                                  float* __restrict__ mmax, int E) {
  int t = blockIdx.x * blockDim.x + threadIdx.x;
  if (t >= E * 2) return;
  int e = t >> 1, h = t & 1;
  int s = src[e], d = dst[e];
  const float* ps = fs + s * 128 + h * 64;
  const float* pd = fd + d * 128 + h * 64;
  const float* pa = attn + h * 64;
  float acc = 0.f;
#pragma unroll
  for (int i = 0; i < 64; i += 4) {
    float4 a = *(const float4*)(ps + i);
    float4 b = *(const float4*)(pd + i);
    float4 w = *(const float4*)(pa + i);
    float e0 = a.x + b.x; e0 = (e0 > 0.f) ? e0 : 0.2f * e0;
    float e1 = a.y + b.y; e1 = (e1 > 0.f) ? e1 : 0.2f * e1;
    float e2 = a.z + b.z; e2 = (e2 > 0.f) ? e2 : 0.2f * e2;
    float e3 = a.w + b.w; e3 = (e3 > 0.f) ? e3 : 0.2f * e3;
    acc += w.x * e0 + w.y * e1 + w.z * e2 + w.w * e3;
  }
  logits[t] = acc;
  atomicMaxF(&mmax[d * 2 + h], acc);
}

__global__ void gat_exp_kernel(float* __restrict__ logits,
                               const float* __restrict__ mmax,
                               float* __restrict__ denom,
                               const int* __restrict__ dst, int E) {
  int t = blockIdx.x * blockDim.x + threadIdx.x;
  if (t >= E * 2) return;
  int e = t >> 1, h = t & 1, d = dst[e];
  float ex = __expf(logits[t] - mmax[d * 2 + h]);
  logits[t] = ex;                 // reuse buffer: now holds exp values
  atomicAdd(&denom[d * 2 + h], ex);
}

__global__ void gat_scatter_kernel(const float* __restrict__ fs,
                                   const float* __restrict__ ex,
                                   const float* __restrict__ denom,
                                   const int* __restrict__ src,
                                   const int* __restrict__ dst,
                                   float* __restrict__ outacc, int E) {
  int t = blockIdx.x * blockDim.x + threadIdx.x;
  if (t >= E * 32) return;
  int e = t >> 5, r = t & 31;
  int h = r >> 4, g = (r & 15) << 2;
  int s = src[e], d = dst[e];
  float alpha = ex[e * 2 + h] / denom[d * 2 + h];
  float4 v = *(const float4*)(fs + s * 128 + h * 64 + g);
  float* a = outacc + d * 128 + h * 64 + g;
  atomicAdd(a + 0, alpha * v.x); atomicAdd(a + 1, alpha * v.y);
  atomicAdd(a + 2, alpha * v.z); atomicAdd(a + 3, alpha * v.w);
}

// out[n,d] = mean_h( gelu(acc[n,h,d] + bias[h,d]) ) * drop
__global__ void gat_finish_kernel(const float* __restrict__ acc,
                                  const float* __restrict__ gbias,
                                  const float* __restrict__ drop,
                                  float* __restrict__ out, int n) {
  int i = blockIdx.x * blockDim.x + threadIdx.x;
  if (i >= n) return;
  int row = i >> 6, d = i & 63;
  float v0 = acc[row * 128 + d]      + gbias[d];
  float v1 = acc[row * 128 + 64 + d] + gbias[64 + d];
  float m  = 0.5f * (gelu_exact(v0) + gelu_exact(v1));
  if (drop) m *= drop[i];
  out[i] = m;
}

// one wave32 per row; lane holds cols {lane, lane+32}; shfl_xor reduce
__global__ void l2norm_kernel(const float* __restrict__ x,
                              float* __restrict__ out, int nRows) {
  int wave = (blockIdx.x * blockDim.x + threadIdx.x) >> 5;
  int lane = threadIdx.x & 31;
  if (wave >= nRows) return;
  const float* px = x + wave * 64;
  float a = px[lane], b = px[lane + 32];
  float ss = a * a + b * b;
#pragma unroll
  for (int off = 16; off; off >>= 1) ss += __shfl_xor(ss, off, 32);
  float scale = 1.0f / fmaxf(sqrtf(ss), 1e-12f);
  out[wave * 64 + lane]      = a * scale;
  out[wave * 64 + lane + 32] = b * scale;
}

// =====================================================================
extern "C" void kernel_launch(void* const* d_in, const int* in_sizes, int n_in,
                              void* d_out, int out_size, void* d_ws, size_t ws_size,
                              hipStream_t stream) {
  (void)in_sizes; (void)n_in; (void)out_size; (void)ws_size;
  const float* feat     = (const float*)d_in[0];
  const int*   esrc     = (const int*)  d_in[1];
  const int*   edst     = (const int*)  d_in[2];
  const float* Wmt      = (const float*)d_in[3];
  const float* bmt      = (const float*)d_in[4];
  const float* Wms      = (const float*)d_in[5];
  const float* bms      = (const float*)d_in[6];
  const float* W1_self  = (const float*)d_in[7];
  const float* W1_neigh = (const float*)d_in[8];
  const float* b1       = (const float*)d_in[9];
  const float* W2_self  = (const float*)d_in[10];
  const float* W2_neigh = (const float*)d_in[11];
  const float* b2       = (const float*)d_in[12];
  const float* g1_Wsrc  = (const float*)d_in[13];
  const float* g1_Wdst  = (const float*)d_in[14];
  const float* g1_attn  = (const float*)d_in[15];
  const float* g1_bias  = (const float*)d_in[16];
  const float* g2_Wsrc  = (const float*)d_in[17];
  const float* g2_Wdst  = (const float*)d_in[18];
  const float* g2_attn  = (const float*)d_in[19];
  const float* g2_bias  = (const float*)d_in[20];
  const float* W3       = (const float*)d_in[21];
  const float* b3       = (const float*)d_in[22];
  const float* drop_t   = (const float*)d_in[23];
  const float* drop_s   = (const float*)d_in[24];

  const int N = N_NODES, E = N_EDGES;

  // ---- workspace carve-up (floats): ~133 MB
  float* ws       = (float*)d_ws;
  float* buf128_a = ws;                       // N*128  gated feats / GAT accum
  float* buf128_b = buf128_a + N * 128;       // N*128  fs
  float* buf128_c = buf128_b + N * 128;       // N*128  fd
  float* Y1       = buf128_c + N * 128;       // N*64
  float* Y2       = Y1 + N * 64;              // N*64
  float* agg      = Y2 + N * 64;              // N*64
  float* hbuf     = agg + N * 64;             // N*64   h1/h2/h_s1
  float* deg      = hbuf + N * 64;            // N
  float* logits   = deg + N;                  // E*2
  float* mmax     = logits + E * 2;           // N*2
  float* denom    = mmax + N * 2;             // N*2

  float* out_t = (float*)d_out;               // [N,64]
  float* out_s = out_t + N * 64;              // [N,64]

  const int GB = 160;                         // GEMM grid: 160 blocks x 8 waves
  const int TB = 256;
  const int gE1  = (E * 2  + TB - 1) / TB;
  const int gE16 = (E * 16 + TB - 1) / TB;
  const int gE32 = (E * 32 + TB - 1) / TB;
  const int gN64 = (N * 64 + TB - 1) / TB;
  const int gE   = (E + TB - 1) / TB;
  const int gN2  = (N * 2 + TB - 1) / TB;
  const int gNorm = (N + 7) / 8;              // 8 waves / block

  // ======================= topology branch (SAGE) =======================
  // h_t0 = feat * sigmoid(feat @ Wmt + bmt)
  wmma_gemm_kernel<4,128,1,true,false><<<GB, TB, 0, stream>>>(
      feat, 128, 128, nullptr, 0, Wmt, bmt, buf128_a, 128, N_ROWTILES, feat, 128);
  hipMemsetAsync(deg, 0, N * sizeof(float), stream);
  deg_kernel<<<gE, TB, 0, stream>>>(edst, deg, E);
  wmma_gemm_kernel<4,64,0,false,false><<<GB, TB, 0, stream>>>(
      buf128_a, 128, 128, nullptr, 0, W1_self, nullptr, Y1, 64, N_ROWTILES, nullptr, 0);
  wmma_gemm_kernel<4,64,0,false,false><<<GB, TB, 0, stream>>>(
      buf128_a, 128, 128, nullptr, 0, W1_neigh, nullptr, Y2, 64, N_ROWTILES, nullptr, 0);
  hipMemsetAsync(agg, 0, N * 64 * sizeof(float), stream);
  scatter_add64_kernel<<<gE16, TB, 0, stream>>>(Y2, esrc, edst, agg, E);
  sage_combine_kernel<<<gN64, TB, 0, stream>>>(Y1, agg, deg, b1, drop_t, hbuf, N * 64);
  wmma_gemm_kernel<2,64,0,false,false><<<GB, TB, 0, stream>>>(
      hbuf, 64, 64, nullptr, 0, W2_self, nullptr, Y1, 64, N_ROWTILES, nullptr, 0);
  wmma_gemm_kernel<2,64,0,false,false><<<GB, TB, 0, stream>>>(
      hbuf, 64, 64, nullptr, 0, W2_neigh, nullptr, Y2, 64, N_ROWTILES, nullptr, 0);
  hipMemsetAsync(agg, 0, N * 64 * sizeof(float), stream);
  scatter_add64_kernel<<<gE16, TB, 0, stream>>>(Y2, esrc, edst, agg, E);
  sage_combine_kernel<<<gN64, TB, 0, stream>>>(Y1, agg, deg, b2, nullptr, hbuf, N * 64);
  // concat([h2, feat]) @ W3 + b3 -> leaky(0.01) -> L2 norm
  wmma_gemm_kernel<6,64,2,true,true><<<GB, TB, 0, stream>>>(
      hbuf, 64, 64, feat, 128, W3, b3, Y1, 64, N_ROWTILES, nullptr, 0);
  l2norm_kernel<<<gNorm, TB, 0, stream>>>(Y1, out_t, N);

  // ======================= semantic branch (GATv2) ======================
  wmma_gemm_kernel<4,128,1,true,false><<<GB, TB, 0, stream>>>(
      feat, 128, 128, nullptr, 0, Wms, bms, buf128_a, 128, N_ROWTILES, feat, 128);
  wmma_gemm_kernel<4,128,0,false,false><<<GB, TB, 0, stream>>>(
      buf128_a, 128, 128, nullptr, 0, g1_Wsrc, nullptr, buf128_b, 128, N_ROWTILES, nullptr, 0);
  wmma_gemm_kernel<4,128,0,false,false><<<GB, TB, 0, stream>>>(
      buf128_a, 128, 128, nullptr, 0, g1_Wdst, nullptr, buf128_c, 128, N_ROWTILES, nullptr, 0);
  fill_kernel<<<gN2, TB, 0, stream>>>(mmax, -3.402823466e38f, N * 2);
  gat_logits_kernel<<<gE1, TB, 0, stream>>>(buf128_b, buf128_c, g1_attn, esrc, edst, logits, mmax, E);
  hipMemsetAsync(denom, 0, N * 2 * sizeof(float), stream);
  gat_exp_kernel<<<gE1, TB, 0, stream>>>(logits, mmax, denom, edst, E);
  hipMemsetAsync(buf128_a, 0, (size_t)N * 128 * sizeof(float), stream);
  gat_scatter_kernel<<<gE32, TB, 0, stream>>>(buf128_b, logits, denom, esrc, edst, buf128_a, E);
  gat_finish_kernel<<<gN64, TB, 0, stream>>>(buf128_a, g1_bias, drop_s, hbuf, N * 64);
  wmma_gemm_kernel<2,128,0,false,false><<<GB, TB, 0, stream>>>(
      hbuf, 64, 64, nullptr, 0, g2_Wsrc, nullptr, buf128_b, 128, N_ROWTILES, nullptr, 0);
  wmma_gemm_kernel<2,128,0,false,false><<<GB, TB, 0, stream>>>(
      hbuf, 64, 64, nullptr, 0, g2_Wdst, nullptr, buf128_c, 128, N_ROWTILES, nullptr, 0);
  fill_kernel<<<gN2, TB, 0, stream>>>(mmax, -3.402823466e38f, N * 2);
  gat_logits_kernel<<<gE1, TB, 0, stream>>>(buf128_b, buf128_c, g2_attn, esrc, edst, logits, mmax, E);
  hipMemsetAsync(denom, 0, N * 2 * sizeof(float), stream);
  gat_exp_kernel<<<gE1, TB, 0, stream>>>(logits, mmax, denom, edst, E);
  hipMemsetAsync(buf128_a, 0, (size_t)N * 128 * sizeof(float), stream);
  gat_scatter_kernel<<<gE32, TB, 0, stream>>>(buf128_b, logits, denom, esrc, edst, buf128_a, E);
  gat_finish_kernel<<<gN64, TB, 0, stream>>>(buf128_a, g2_bias, nullptr, Y1, N * 64);
  l2norm_kernel<<<gNorm, TB, 0, stream>>>(Y1, out_s, N);
}
